// RingAttentionLayerSim_343597384114
// MI455X (gfx1250) — matrix operand order repaired and checked by reading, not compiled, not run on hardware
//
#include <hip/hip_runtime.h>
#include <stdint.h>

#define HIDDEN 2048
#define SEQ    2048
#define NHEADS 16
#define HDIM   128

typedef __attribute__((ext_vector_type(16))) __bf16   v16bf;
typedef __attribute__((ext_vector_type(8)))  float    v8f;
typedef __attribute__((ext_vector_type(4)))  uint32_t u32x4;
typedef __attribute__((ext_vector_type(4)))  float    f32x4;

struct Pair16 { u32x4 lo; u32x4 hi; };
union FragCvt { Pair16 p; v16bf v; };

static __device__ inline v16bf make_frag(u32x4 lo, u32x4 hi) {
  FragCvt f; f.p.lo = lo; f.p.hi = hi; return f.v;
}

// round-to-nearest-even float -> bf16 (bit pattern), matches XLA conversion
static __device__ inline uint32_t bf16_1(float f) {
  uint32_t x = __builtin_bit_cast(uint32_t, f);
  x += 0x7fffu + ((x >> 16) & 1u);
  return x >> 16;
}
static __device__ inline uint32_t bf16_pk(float a, float b) {
  return bf16_1(a) | (bf16_1(b) << 16);
}

static __device__ inline v8f zero8() {
  v8f z;
#pragma unroll
  for (int i = 0; i < 8; ++i) z[i] = 0.0f;
  return z;
}

// ---- CDNA5 async global->LDS copy engine (ASYNCcnt-tracked) -----------------
// Per ISA 10.x aperture rules the low 32 bits of a generic __shared__ pointer
// are the LDS byte address, which is what VDST of the async load expects.
static __device__ inline void async_copy_b128(void* lds_ptr, const void* gptr) {
  const uint32_t lds_addr = (uint32_t)(uintptr_t)lds_ptr;
  asm volatile("global_load_async_to_lds_b128 %0, %1, off"
               :: "v"(lds_addr), "v"((uint64_t)(uintptr_t)gptr)
               : "memory");
}
static __device__ inline void wait_async0() {
  asm volatile("s_wait_asynccnt 0x0" ::: "memory");
}

// ---------------------------------------------------------------------------
// C[M,N] = A[M,K] * W[N,K]^T + bias[N]   (nn.Linear semantics), M=N=K=2048
// A: fp32 or bf16 (row-major), W: fp32 row-major [N][K], C: fp32 or bf16
// Tile: 128x128 per workgroup, BK=32, 8 waves, wave owns 16 rows x 128 cols.
// Double-buffered LDS: global loads for step kt+1 are in flight during the
// WMMA stream of step kt; conversion+LDS store happens after compute.
// ---------------------------------------------------------------------------
template <bool IN_F32, bool OUT_F32>
__global__ __launch_bounds__(256) void gemm_nt_bias(
    const void* __restrict__ Ain, const float* __restrict__ W,
    const float* __restrict__ bias, void* __restrict__ Cout) {
  __shared__ __align__(16) uint16_t As[2][128 * 32];  // [buf][row][k] bf16
  __shared__ __align__(16) uint16_t Ws[2][128 * 32];  // [buf][n]  [k] bf16

  const int tid  = threadIdx.x;
  const int lane = tid & 31;
  const int wave = tid >> 5;
  const int half = lane >> 4;   // 0/1: lane group
  const int l16  = lane & 15;

  const int bm = (int)blockIdx.x * 128;
  const int bn = (int)blockIdx.y * 128;

  const int lrow = tid >> 1;         // 0..127 tile row loaded by this thread
  const int lcol = (tid & 1) << 4;   // 0 or 16

  const int KTILES = HIDDEN / 32;

  v8f acc[8];
#pragma unroll
  for (int i = 0; i < 8; ++i) acc[i] = zero8();

  f32x4 arF[4], wrF[4];
  u32x4 arB[2];

  // ---- prologue: stage tile 0 into buffer 0 ----
  {
    if (IN_F32) {
      const float* ap = (const float*)Ain + (size_t)(bm + lrow) * HIDDEN + lcol;
      u32x4 lo = { bf16_pk(ap[0], ap[1]), bf16_pk(ap[2], ap[3]),
                   bf16_pk(ap[4], ap[5]), bf16_pk(ap[6], ap[7]) };
      u32x4 hi = { bf16_pk(ap[8], ap[9]),  bf16_pk(ap[10], ap[11]),
                   bf16_pk(ap[12], ap[13]), bf16_pk(ap[14], ap[15]) };
      *(u32x4*)&As[0][lrow * 32 + lcol]     = lo;
      *(u32x4*)&As[0][lrow * 32 + lcol + 8] = hi;
    } else {
      const uint16_t* ap = (const uint16_t*)Ain + (size_t)(bm + lrow) * HIDDEN + lcol;
      *(u32x4*)&As[0][lrow * 32 + lcol]     = ((const u32x4*)ap)[0];
      *(u32x4*)&As[0][lrow * 32 + lcol + 8] = ((const u32x4*)ap)[1];
    }
    const float* wp = W + (size_t)(bn + lrow) * HIDDEN + lcol;
    u32x4 lo = { bf16_pk(wp[0], wp[1]), bf16_pk(wp[2], wp[3]),
                 bf16_pk(wp[4], wp[5]), bf16_pk(wp[6], wp[7]) };
    u32x4 hi = { bf16_pk(wp[8], wp[9]),  bf16_pk(wp[10], wp[11]),
                 bf16_pk(wp[12], wp[13]), bf16_pk(wp[14], wp[15]) };
    *(u32x4*)&Ws[0][lrow * 32 + lcol]     = lo;
    *(u32x4*)&Ws[0][lrow * 32 + lcol + 8] = hi;
  }
  __syncthreads();

  for (int kt = 0; kt < KTILES; ++kt) {
    const int cur = kt & 1;

    // ---- issue global loads for tile kt+1 (raw, convert later) ----
    if (kt + 1 < KTILES) {
      const int k0 = (kt + 1) * 32;
      if (IN_F32) {
        const float* ap = (const float*)Ain + (size_t)(bm + lrow) * HIDDEN + k0 + lcol;
        arF[0] = ((const f32x4*)ap)[0];
        arF[1] = ((const f32x4*)ap)[1];
        arF[2] = ((const f32x4*)ap)[2];
        arF[3] = ((const f32x4*)ap)[3];
      } else {
        const uint16_t* ap = (const uint16_t*)Ain + (size_t)(bm + lrow) * HIDDEN + k0 + lcol;
        arB[0] = ((const u32x4*)ap)[0];
        arB[1] = ((const u32x4*)ap)[1];
      }
      const float* wp = W + (size_t)(bn + lrow) * HIDDEN + k0 + lcol;
      wrF[0] = ((const f32x4*)wp)[0];
      wrF[1] = ((const f32x4*)wp)[1];
      wrF[2] = ((const f32x4*)wp)[2];
      wrF[3] = ((const f32x4*)wp)[3];
    }
    // ---- prefetch tile kt+2 toward L2 (global_prefetch_b8) ----
    if (kt + 2 < KTILES) {
      const int k2 = (kt + 2) * 32;
      const size_t aoff = ((size_t)(bm + lrow) * HIDDEN + k2 + lcol) * (IN_F32 ? 4 : 2);
      __builtin_prefetch((const char*)Ain + aoff, 0, 3);
      __builtin_prefetch(W + (size_t)(bn + lrow) * HIDDEN + k2 + lcol, 0, 3);
    }

    // ---- WMMA compute from buffer `cur` ----
    {
      const int arow = wave * 16 + l16;
      const uint16_t* abase = &As[cur][arow * 32 + half * 8];
      v16bf afrag = make_frag(*(const u32x4*)abase, *(const u32x4*)(abase + 16));
#pragma unroll
      for (int nt = 0; nt < 8; ++nt) {
        const uint16_t* bbase = &Ws[cur][(nt * 16 + l16) * 32 + half * 8];
        v16bf bfrag = make_frag(*(const u32x4*)bbase, *(const u32x4*)(bbase + 16));
        acc[nt] = __builtin_amdgcn_wmma_f32_16x16x32_bf16(
            false, afrag, false, bfrag, (short)0, acc[nt], false, false);
      }
    }

    // ---- convert + store tile kt+1 into the other buffer ----
    if (kt + 1 < KTILES) {
      const int nxt = cur ^ 1;
      if (IN_F32) {
        u32x4 lo = { bf16_pk(arF[0].x, arF[0].y), bf16_pk(arF[0].z, arF[0].w),
                     bf16_pk(arF[1].x, arF[1].y), bf16_pk(arF[1].z, arF[1].w) };
        u32x4 hi = { bf16_pk(arF[2].x, arF[2].y), bf16_pk(arF[2].z, arF[2].w),
                     bf16_pk(arF[3].x, arF[3].y), bf16_pk(arF[3].z, arF[3].w) };
        *(u32x4*)&As[nxt][lrow * 32 + lcol]     = lo;
        *(u32x4*)&As[nxt][lrow * 32 + lcol + 8] = hi;
      } else {
        *(u32x4*)&As[nxt][lrow * 32 + lcol]     = arB[0];
        *(u32x4*)&As[nxt][lrow * 32 + lcol + 8] = arB[1];
      }
      u32x4 lo = { bf16_pk(wrF[0].x, wrF[0].y), bf16_pk(wrF[0].z, wrF[0].w),
                   bf16_pk(wrF[1].x, wrF[1].y), bf16_pk(wrF[1].z, wrF[1].w) };
      u32x4 hi = { bf16_pk(wrF[2].x, wrF[2].y), bf16_pk(wrF[2].z, wrF[2].w),
                   bf16_pk(wrF[3].x, wrF[3].y), bf16_pk(wrF[3].z, wrF[3].w) };
      *(u32x4*)&Ws[nxt][lrow * 32 + lcol]     = lo;
      *(u32x4*)&Ws[nxt][lrow * 32 + lcol + 8] = hi;
    }
    __syncthreads();
  }

  // ---- epilogue: C layout row m = r + 8*half, col n = lane%16 ----
#pragma unroll
  for (int nt = 0; nt < 8; ++nt) {
    const int col = bn + nt * 16 + l16;
    const float bv = bias[col];
#pragma unroll
    for (int r = 0; r < 8; ++r) {
      const int row = bm + wave * 16 + r + 8 * half;
      const float v = acc[nt][r] + bv;
      if (OUT_F32)
        ((float*)Cout)[(size_t)row * HIDDEN + col] = v;
      else
        ((uint16_t*)Cout)[(size_t)row * HIDDEN + col] = (uint16_t)bf16_1(v);
    }
  }
}

// ---------------------------------------------------------------------------
// Flash attention, one head x 128 query rows per workgroup.
// Ring with WORLD_SIZE identical KV shards == plain softmax attention.
// Q,K,V,O: bf16 [SEQ][HIDDEN], head h occupies columns h*128..h*128+127.
// K staging uses the CDNA5 async global->LDS engine into the inactive double
// buffer; V is staged through registers (transposed on the way into LDS).
// ---------------------------------------------------------------------------
__global__ __launch_bounds__(256) void flash_attn(
    const uint16_t* __restrict__ Q, const uint16_t* __restrict__ K,
    const uint16_t* __restrict__ V, uint16_t* __restrict__ O) {
  __shared__ __align__(16) uint16_t Kb[2][64 * 128];   // [buf][key][dim]
  __shared__ __align__(16) uint16_t Vt[2][128 * 64];   // [buf][dim][key]
  __shared__ __align__(16) uint16_t Pb[8][16 * 64];    // per-wave P tile

  const int tid  = threadIdx.x;
  const int lane = tid & 31;
  const int wave = tid >> 5;
  const int half = lane >> 4;
  const int l16  = lane & 15;

  const int head = blockIdx.y;
  const int qblk = blockIdx.x;
  const int hcol = head * HDIM;
  const int qrow = qblk * 128 + wave * 16 + l16;  // A-fragment row of this lane

  const int NB = SEQ / 64;

  // persistent Q fragments: head_dim 128 = 4 k-steps of 32
  v16bf qf[4];
#pragma unroll
  for (int ks = 0; ks < 4; ++ks) {
    const uint16_t* qp = Q + (size_t)qrow * HIDDEN + hcol + ks * 32 + half * 8;
    qf[ks] = make_frag(*(const u32x4*)qp, *(const u32x4*)(qp + 16));
  }

  v8f out[8];
#pragma unroll
  for (int i = 0; i < 8; ++i) out[i] = zero8();
  float mst[8], lsum[8];
#pragma unroll
  for (int r = 0; r < 8; ++r) { mst[r] = -3.0e38f; lsum[r] = 0.0f; }

  const float scale = 0.08838834764831845f;  // 1/sqrt(128)

  const int krow = tid >> 2;         // 0..63: key loaded by this thread
  const int kcol = (tid & 3) << 5;   // 0,32,64,96: dim base (32 elems)

  union VRegs { u32x4 q[4]; uint32_t w[16]; } vr;

  // ---- prologue: stage block 0 into buffer 0 ----
  {
    const uint16_t* ksrc = K + (size_t)krow * HIDDEN + hcol + kcol;
    uint16_t* kdst = &Kb[0][krow * 128 + kcol];
    async_copy_b128(kdst +  0, ksrc +  0);
    async_copy_b128(kdst +  8, ksrc +  8);
    async_copy_b128(kdst + 16, ksrc + 16);
    async_copy_b128(kdst + 24, ksrc + 24);

    const uint16_t* vsrc = V + (size_t)krow * HIDDEN + hcol + kcol;
    VRegs v0;
    v0.q[0] = ((const u32x4*)vsrc)[0];
    v0.q[1] = ((const u32x4*)vsrc)[1];
    v0.q[2] = ((const u32x4*)vsrc)[2];
    v0.q[3] = ((const u32x4*)vsrc)[3];
#pragma unroll
    for (int e = 0; e < 16; ++e) {
      const uint32_t wv = v0.w[e];
      const int d = kcol + e * 2;
      Vt[0][(d + 0) * 64 + krow] = (uint16_t)(wv & 0xffffu);
      Vt[0][(d + 1) * 64 + krow] = (uint16_t)(wv >> 16);
    }
    wait_async0();
  }
  __syncthreads();

  for (int kb = 0; kb < NB; ++kb) {
    const int cur = kb & 1;
    const int nxt = cur ^ 1;

    // ---- stage block kb+1: async K copy into inactive buffer, V into regs ----
    if (kb + 1 < NB) {
      const uint16_t* ksrc = K + (size_t)((kb + 1) * 64 + krow) * HIDDEN + hcol + kcol;
      uint16_t* kdst = &Kb[nxt][krow * 128 + kcol];
      async_copy_b128(kdst +  0, ksrc +  0);
      async_copy_b128(kdst +  8, ksrc +  8);
      async_copy_b128(kdst + 16, ksrc + 16);
      async_copy_b128(kdst + 24, ksrc + 24);

      const uint16_t* vsrc = V + (size_t)((kb + 1) * 64 + krow) * HIDDEN + hcol + kcol;
      vr.q[0] = ((const u32x4*)vsrc)[0];
      vr.q[1] = ((const u32x4*)vsrc)[1];
      vr.q[2] = ((const u32x4*)vsrc)[2];
      vr.q[3] = ((const u32x4*)vsrc)[3];
    }
    if (kb + 2 < NB) {
      __builtin_prefetch(K + (size_t)((kb + 2) * 64 + krow) * HIDDEN + hcol + kcol, 0, 3);
      __builtin_prefetch(V + (size_t)((kb + 2) * 64 + krow) * HIDDEN + hcol + kcol, 0, 3);
    }

    // ---- scores: S(16x64) = Q(16x128) * K^T(128x64), 4 n-tiles x 4 k-steps ----
    v8f sc[4];
#pragma unroll
    for (int nt = 0; nt < 4; ++nt) {
      sc[nt] = zero8();
#pragma unroll
      for (int ks = 0; ks < 4; ++ks) {
        const uint16_t* kp = &Kb[cur][(nt * 16 + l16) * 128 + ks * 32 + half * 8];
        v16bf bfrag = make_frag(*(const u32x4*)kp, *(const u32x4*)(kp + 16));
        sc[nt] = __builtin_amdgcn_wmma_f32_16x16x32_bf16(
            false, qf[ks], false, bfrag, (short)0, sc[nt], false, false);
      }
    }

    // ---- online softmax (rows = r + 8*half, cols across 16 lanes) ----
    float rmax[8];
#pragma unroll
    for (int r = 0; r < 8; ++r) {
#pragma unroll
      for (int nt = 0; nt < 4; ++nt) sc[nt][r] *= scale;
      float m = fmaxf(fmaxf(sc[0][r], sc[1][r]), fmaxf(sc[2][r], sc[3][r]));
      m = fmaxf(m, __shfl_xor(m, 1));
      m = fmaxf(m, __shfl_xor(m, 2));
      m = fmaxf(m, __shfl_xor(m, 4));
      m = fmaxf(m, __shfl_xor(m, 8));
      rmax[r] = m;
    }

    float mnew[8], oscale[8], rsum[8];
#pragma unroll
    for (int r = 0; r < 8; ++r) {
      mnew[r]   = fmaxf(mst[r], rmax[r]);
      oscale[r] = __expf(mst[r] - mnew[r]);
      rsum[r]   = 0.0f;
    }

    // P = exp(S - mnew) -> bf16 into per-wave LDS buffer, accumulate row sums
#pragma unroll
    for (int nt = 0; nt < 4; ++nt) {
#pragma unroll
      for (int r = 0; r < 8; ++r) {
        const float p = __expf(sc[nt][r] - mnew[r]);
        rsum[r] += p;
        Pb[wave][(r + 8 * half) * 64 + nt * 16 + l16] = (uint16_t)bf16_1(p);
      }
    }

#pragma unroll
    for (int r = 0; r < 8; ++r) {
      float s = rsum[r];
      s += __shfl_xor(s, 1);
      s += __shfl_xor(s, 2);
      s += __shfl_xor(s, 4);
      s += __shfl_xor(s, 8);
      lsum[r] = lsum[r] * oscale[r] + s;
      mst[r]  = mnew[r];
    }

    // rescale running output, then out += P(16x64) * V(64x128)
#pragma unroll
    for (int nt = 0; nt < 8; ++nt)
#pragma unroll
      for (int r = 0; r < 8; ++r) out[nt][r] *= oscale[r];

#pragma unroll
    for (int ks = 0; ks < 2; ++ks) {
      const uint16_t* pp = &Pb[wave][l16 * 64 + ks * 32 + half * 8];
      v16bf afrag = make_frag(*(const u32x4*)pp, *(const u32x4*)(pp + 16));
#pragma unroll
      for (int nt = 0; nt < 8; ++nt) {
        const uint16_t* vp = &Vt[cur][(nt * 16 + l16) * 64 + ks * 32 + half * 8];
        v16bf bfrag = make_frag(*(const u32x4*)vp, *(const u32x4*)(vp + 16));
        out[nt] = __builtin_amdgcn_wmma_f32_16x16x32_bf16(
            false, afrag, false, bfrag, (short)0, out[nt], false, false);
      }
    }

    // ---- store staged V block kb+1 (transposed) into the other buffer ----
    if (kb + 1 < NB) {
#pragma unroll
      for (int e = 0; e < 16; ++e) {
        const uint32_t wv = vr.w[e];
        const int d = kcol + e * 2;
        Vt[nxt][(d + 0) * 64 + krow] = (uint16_t)(wv & 0xffffu);
        Vt[nxt][(d + 1) * 64 + krow] = (uint16_t)(wv >> 16);
      }
    }
    wait_async0();   // async K copy for kb+1 complete before the barrier
    __syncthreads();
  }

  // ---- finalize: divide by softmax sum, store bf16 ----
#pragma unroll
  for (int nt = 0; nt < 8; ++nt) {
#pragma unroll
    for (int r = 0; r < 8; ++r) {
      const int row = qblk * 128 + wave * 16 + r + 8 * half;
      const float v = out[nt][r] / lsum[r];
      O[(size_t)row * HIDDEN + hcol + nt * 16 + l16] = (uint16_t)bf16_1(v);
    }
  }
}

// ---------------------------------------------------------------------------
extern "C" void kernel_launch(void* const* d_in, const int* in_sizes, int n_in,
                              void* d_out, int out_size, void* d_ws, size_t ws_size,
                              hipStream_t stream) {
  (void)in_sizes; (void)n_in; (void)out_size; (void)ws_size;
  const float* hs = (const float*)d_in[0];
  const float* wq = (const float*)d_in[1];
  const float* bq = (const float*)d_in[2];
  const float* wk = (const float*)d_in[3];
  const float* bk = (const float*)d_in[4];
  const float* wv = (const float*)d_in[5];
  const float* bv = (const float*)d_in[6];
  const float* wo = (const float*)d_in[7];
  const float* bo = (const float*)d_in[8];

  uint16_t* Qw = (uint16_t*)d_ws;
  uint16_t* Kw = Qw + (size_t)SEQ * HIDDEN;
  uint16_t* Vw = Kw + (size_t)SEQ * HIDDEN;
  uint16_t* Aw = Vw + (size_t)SEQ * HIDDEN;

  const dim3 grid(16, 16), block(256);
  gemm_nt_bias<true, false><<<grid, block, 0, stream>>>(hs, wq, bq, Qw);
  gemm_nt_bias<true, false><<<grid, block, 0, stream>>>(hs, wk, bk, Kw);
  gemm_nt_bias<true, false><<<grid, block, 0, stream>>>(hs, wv, bv, Vw);
  flash_attn<<<dim3(16, 16), block, 0, stream>>>(Qw, Kw, Vw, Aw);
  gemm_nt_bias<false, true><<<grid, block, 0, stream>>>(Aw, wo, bo, (float*)d_out);
}
